// BiaffineCRFClassifier_39805756899334
// MI455X (gfx1250) — matrix operand order repaired and checked by reading, not compiled, not run on hardware
//
#include <hip/hip_runtime.h>
#include <hip/hip_bf16.h>

// ---------------------------------------------------------------------------
// BiaffineCRFClassifier for MI455X (gfx1250, wave32, WMMA).
// GEMMs: 128x64 workgroup tile, 8 waves in 4x2, each wave 32x32 C (4 WMMAs
// per K-step of v_wmma_f32_16x16x32_f16).
//   GEMM-1 (MLP): regular b128 LDS staging + global_prefetch_b8
//   GEMM-2 (bilin1): B block loaded ONCE per workgroup via TDM
//                    tensor_load_to_lds + s_wait_tensorcnt (TENSORcnt path)
//   GEMM-3 (bilin2): per-K-step staging via global_load_async_to_lds_b128 +
//                    s_wait_asynccnt (ASYNCcnt path)
// ---------------------------------------------------------------------------

typedef __attribute__((ext_vector_type(16))) _Float16 v16h;
typedef __attribute__((ext_vector_type(8)))  _Float16 h8;
typedef __attribute__((ext_vector_type(8)))  float    v8f;
typedef __attribute__((ext_vector_type(4)))  unsigned int ui32x4;
typedef __attribute__((ext_vector_type(8)))  int      i32x8;
typedef __attribute__((ext_vector_type(4)))  int      i32x4;
typedef unsigned char u8;

#define Bsz   16
#define Nseq  256
#define Hdim  1024
#define HIDc  300
#define Lnum  64
#define PADW  320              // HID+1=301 padded to 320
#define MROWS 4096             // B*N
#define LDST  40               // LDS tile row stride in halves (32 + 8 pad)

union F16x16 { v16h v; h8 h[2]; };

#define WMMA_F16(a, b, c) \
  __builtin_amdgcn_wmma_f32_16x16x32_f16(false, (a), false, (b), (short)0, (c), false, false)

// A fragment (16x32 f16): lanes0-15 hold K 0..7 & 16..23, lanes16-31 hold 8..15 & 24..31
__device__ __forceinline__ v16h load_a_frag(const _Float16* as, int lane) {
  int mrow = lane & 15;
  int kb = (lane & 16) ? 8 : 0;
  F16x16 f;
  f.h[0] = *reinterpret_cast<const h8*>(as + mrow * LDST + kb);
  f.h[1] = *reinterpret_cast<const h8*>(as + mrow * LDST + kb + 16);
  return f.v;
}
// B fragment (32x16 f16) from LDS tile stored transposed BT[n][k], stride LDST
__device__ __forceinline__ v16h load_b_frag(const _Float16* bs, int lane) {
  int nrow = lane & 15;
  int kb = (lane & 16) ? 8 : 0;
  F16x16 f;
  f.h[0] = *reinterpret_cast<const h8*>(bs + nrow * LDST + kb);
  f.h[1] = *reinterpret_cast<const h8*>(bs + nrow * LDST + kb + 16);
  return f.v;
}
// Same but with arbitrary row stride (for full-width resident B blocks)
__device__ __forceinline__ v16h load_b_frag_s(const _Float16* bs, int lane, int stride) {
  int nrow = lane & 15;
  int kb = (lane & 16) ? 8 : 0;
  F16x16 f;
  f.h[0] = *reinterpret_cast<const h8*>(bs + nrow * stride + kb);
  f.h[1] = *reinterpret_cast<const h8*>(bs + nrow * stride + kb + 16);
  return f.v;
}

// ---- shared 128x64 GEMM inner loop: C[128x64] += A[128xK] * BT[64xK]^T ----
template <bool ASYNC>
__device__ __forceinline__ void gemm_tile_128x64(
    const _Float16* __restrict__ Ag, int lda,
    const _Float16* __restrict__ Bg, int ldb, int K,
    _Float16* sA, _Float16* sBT, v8f* acc) {
  int tid = threadIdx.x, lane = tid & 31, wave = tid >> 5;
  int mw = wave >> 1, nw = wave & 1;
  int r = tid >> 2, c = (tid & 3) << 3;
  const _Float16* a0p = Ag + (size_t)r * lda + c;
  const _Float16* a1p = Ag + (size_t)(r + 64) * lda + c;
  const _Float16* bp  = Bg + (size_t)r * ldb + c;
  _Float16* sa0 = sA + r * LDST + c;
  _Float16* sa1 = sA + (r + 64) * LDST + c;
  _Float16* sb  = sBT + r * LDST + c;
  const _Float16* fA = sA + (mw * 32) * LDST;
  const _Float16* fB = sBT + (nw * 32) * LDST;
#if defined(__gfx1250__)
  // low 32 bits of a flat LDS address are the in-wave LDS byte offset
  unsigned lo0 = (unsigned)(size_t)sa0;
  unsigned lo1 = (unsigned)(size_t)sa1;
  unsigned lob = (unsigned)(size_t)sb;
#endif
  for (int k0 = 0; k0 < K; k0 += 32) {
    __syncthreads();
    if constexpr (ASYNC) {
#if defined(__gfx1250__)
      asm volatile("global_load_async_to_lds_b128 %0, %1, off"
                   :: "v"(lo0), "v"(a0p + k0) : "memory");
      asm volatile("global_load_async_to_lds_b128 %0, %1, off"
                   :: "v"(lo1), "v"(a1p + k0) : "memory");
      asm volatile("global_load_async_to_lds_b128 %0, %1, off"
                   :: "v"(lob), "v"(bp + k0) : "memory");
      asm volatile("s_wait_asynccnt 0" ::: "memory");
#endif
    } else {
      *reinterpret_cast<h8*>(sa0) = *reinterpret_cast<const h8*>(a0p + k0);
      *reinterpret_cast<h8*>(sa1) = *reinterpret_cast<const h8*>(a1p + k0);
      *reinterpret_cast<h8*>(sb)  = *reinterpret_cast<const h8*>(bp + k0);
      if (k0 + 32 < K) {
        __builtin_prefetch(a0p + k0 + 32, 0, 1);
        __builtin_prefetch(bp + k0 + 32, 0, 1);
      }
    }
    __syncthreads();
    v16h A0 = load_a_frag(fA, lane);
    v16h A1 = load_a_frag(fA + 16 * LDST, lane);
    v16h B0 = load_b_frag(fB, lane);
    v16h B1 = load_b_frag(fB + 16 * LDST, lane);
    acc[0] = WMMA_F16(A0, B0, acc[0]);
    acc[1] = WMMA_F16(A0, B1, acc[1]);
    acc[2] = WMMA_F16(A1, B0, acc[2]);
    acc[3] = WMMA_F16(A1, B1, acc[3]);
  }
}

// -------------------------- prep kernels -----------------------------------
__global__ void k_cast_input(const float* __restrict__ in, _Float16* __restrict__ out, int n) {
  int i = blockIdx.x * 256 + threadIdx.x;
  if (i < n) out[i] = (_Float16)in[i];
}

// BmatT[z][n][k] = W_z[n][k] (n<300), else 0 ; bias[z][n]
__global__ void k_prep_bmatT(const float* __restrict__ Wh, const float* __restrict__ bh,
                             const float* __restrict__ Wd, const float* __restrict__ bd,
                             _Float16* __restrict__ bmatT, float* __restrict__ bias) {
  int idx = blockIdx.x * 256 + threadIdx.x;          // 2*320*1024 exact
  int z = idx / (PADW * Hdim);
  int rem = idx - z * (PADW * Hdim);
  int n = rem / Hdim;
  int k = rem - n * Hdim;
  const float* W  = z ? Wd : Wh;
  const float* bb = z ? bd : bh;
  bmatT[idx] = (_Float16)((n < HIDc) ? W[n * Hdim + k] : 0.f);
  if (k == 0) bias[z * PADW + n] = (n < HIDc) ? bb[n] : 0.f;
}

// WpT[l][j][i] = bilinear_W[l][i][j], padded 301->320
__global__ void k_prep_wpT(const float* __restrict__ bw, _Float16* __restrict__ wpT) {
  int idx = blockIdx.x * 256 + threadIdx.x;          // 64*320*320 exact
  int l = idx / (PADW * PADW);
  int rem = idx - l * (PADW * PADW);
  int j = rem / PADW;
  int i = rem - j * PADW;
  float v = (i <= HIDc && j <= HIDc) ? bw[((size_t)l * 301 + i) * 301 + j] : 0.f;
  wpT[idx] = (_Float16)v;
}

// -------------------------- GEMM 1: fused MLP ------------------------------
__global__ void k_mlp_gemm(const _Float16* __restrict__ A,      // [4096,1024]
                           const _Float16* __restrict__ BmatT,  // [2][320,1024]
                           const float*    __restrict__ bias,   // [2][320]
                           _Float16* __restrict__ h1, _Float16* __restrict__ d1) {
  __shared__ __attribute__((aligned(16))) _Float16 sA[128 * LDST];
  __shared__ __attribute__((aligned(16))) _Float16 sBT[64 * LDST];
  int lane = threadIdx.x & 31, wave = threadIdx.x >> 5;
  int mw = wave >> 1, nw = wave & 1;
  int m0 = blockIdx.x * 128, n0 = blockIdx.y * 64, z = blockIdx.z;
  v8f acc[4] = {};
  gemm_tile_128x64<false>(A + (size_t)m0 * Hdim, Hdim,
                          BmatT + (size_t)z * PADW * Hdim + (size_t)n0 * Hdim,
                          Hdim, Hdim, sA, sBT, acc);
  _Float16* out = z ? d1 : h1;
  int roff = (lane & 16) ? 8 : 0;
#pragma unroll
  for (int si = 0; si < 2; ++si)
#pragma unroll
    for (int sj = 0; sj < 2; ++sj) {
      v8f cacc = acc[si * 2 + sj];
      int col = n0 + nw * 32 + sj * 16 + (lane & 15);
      int rowb = m0 + mw * 32 + si * 16 + roff;
      float bcol = bias[z * PADW + col];
#pragma unroll
      for (int vr = 0; vr < 8; ++vr) {
        float v = fmaxf(cacc[vr] + bcol, 0.f);
        if (col == HIDc) v = 1.f;
        else if (col > HIDc) v = 0.f;
        out[(size_t)(rowb + vr) * PADW + col] = (_Float16)v;
      }
    }
}

// ---------------- GEMM 2: tmp[m,l,j] = sum_i d1[m,i] * Wp[l,i,j] -----------
// B operand (64x320 f16 slice of WpT[l], reused by all K-steps) is loaded
// once per workgroup with a TDM 2D tensor_load_to_lds descriptor.
__global__ void k_bilin1(const _Float16* __restrict__ d1,   // [4096,320]
                         const _Float16* __restrict__ wpT,  // [64][320(j)][320(i)]
                         _Float16* __restrict__ tmp) {      // [4096,64,320]
  __shared__ __attribute__((aligned(16))) _Float16 sA[128 * LDST];
  __shared__ __attribute__((aligned(16))) _Float16 sBfull[64 * PADW];  // 40 KB
  int tid = threadIdx.x, lane = tid & 31, wave = tid >> 5;
  int mw = wave >> 1, nw = wave & 1;
  int m0 = blockIdx.x * 128, j0 = blockIdx.y * 64, l = blockIdx.z;
  const _Float16* bsrc = wpT + (size_t)l * PADW * PADW + (size_t)j0 * PADW;

#if defined(__gfx1250__)
  if (tid < 32) {  // one wave issues the TDM descriptor (EXEC ignored by TDM)
    unsigned long long ga = (unsigned long long)(size_t)bsrc;
    unsigned ldsb = (unsigned)(size_t)sBfull;     // low 32 bits = LDS offset
    // D# group0: count=1 | lds_addr | global_addr | type=2
    ui32x4 g0 = { 1u, ldsb, (unsigned)(ga & 0xFFFFFFFFu),
                  (unsigned)(((ga >> 32) & 0x01FFFFFFu) | (2u << 30)) };
    // D# group1: data_size=2B ; tensor_dim0=320 ; tensor_dim1=64 ;
    //            tile_dim0=320 ; tile_dim1=64 ; tensor_dim0_stride=320
    i32x8 g1 = { (int)(1u << 16),
                 (int)(((unsigned)PADW) << 16),
                 (int)(64u << 16),
                 (int)(((unsigned)PADW) << 16),
                 64,
                 PADW,
                 0, 0 };
    i32x4 gz4 = { 0, 0, 0, 0 };
    i32x8 gz8 = { 0, 0, 0, 0, 0, 0, 0, 0 };
    __builtin_amdgcn_tensor_load_to_lds(g0, g1, gz4, gz4, gz8, 0);
    __builtin_amdgcn_s_wait_tensorcnt(0);
  }
#else
  for (int i = tid * 8; i < 64 * PADW; i += 256 * 8)
    *reinterpret_cast<h8*>(&sBfull[i]) = *reinterpret_cast<const h8*>(bsrc + (i / PADW) * PADW + (i % PADW));
#endif
  __syncthreads();

  v8f acc[4] = {};
  int r = tid >> 2, c = (tid & 3) << 3;
  const _Float16* a0p = d1 + (size_t)(m0 + r) * PADW + c;
  const _Float16* a1p = d1 + (size_t)(m0 + r + 64) * PADW + c;
  _Float16* sa0 = sA + r * LDST + c;
  _Float16* sa1 = sA + (r + 64) * LDST + c;
  const _Float16* fA = sA + (mw * 32) * LDST;
  const _Float16* fB = sBfull + (nw * 32) * PADW;
  for (int k0 = 0; k0 < PADW; k0 += 32) {
    __syncthreads();
    *reinterpret_cast<h8*>(sa0) = *reinterpret_cast<const h8*>(a0p + k0);
    *reinterpret_cast<h8*>(sa1) = *reinterpret_cast<const h8*>(a1p + k0);
    if (k0 + 32 < PADW) __builtin_prefetch(a0p + k0 + 32, 0, 1);
    __syncthreads();
    v16h A0 = load_a_frag(fA, lane);
    v16h A1 = load_a_frag(fA + 16 * LDST, lane);
    v16h B0 = load_b_frag_s(fB + k0, lane, PADW);
    v16h B1 = load_b_frag_s(fB + 16 * PADW + k0, lane, PADW);
    acc[0] = WMMA_F16(A0, B0, acc[0]);
    acc[1] = WMMA_F16(A0, B1, acc[1]);
    acc[2] = WMMA_F16(A1, B0, acc[2]);
    acc[3] = WMMA_F16(A1, B1, acc[3]);
  }
  int roff = (lane & 16) ? 8 : 0;
#pragma unroll
  for (int si = 0; si < 2; ++si)
#pragma unroll
    for (int sj = 0; sj < 2; ++sj) {
      v8f cacc = acc[si * 2 + sj];
      int col = j0 + nw * 32 + sj * 16 + (lane & 15);
      int rowb = m0 + mw * 32 + si * 16 + roff;
#pragma unroll
      for (int vr = 0; vr < 8; ++vr)
        tmp[((size_t)(rowb + vr) * Lnum + l) * PADW + col] = (_Float16)cacc[vr];
    }
}

// ---------- GEMM 3: s[m,y,l] = sum_j tmp[m,l,j] * h1[b,y,j] ----------------
// Uses ASYNC LDS staging (global_load_async_to_lds_b128 + s_wait_asynccnt).
__global__ void k_bilin2(const _Float16* __restrict__ tmp,
                         const _Float16* __restrict__ h1,
                         float* __restrict__ sflat) {       // [4096,256,64] f32
  __shared__ __attribute__((aligned(16))) _Float16 sA[128 * LDST];
  __shared__ __attribute__((aligned(16))) _Float16 sBT[64 * LDST];
  int lane = threadIdx.x & 31, wave = threadIdx.x >> 5;
  int mw = wave >> 1, nw = wave & 1;
  int r0 = blockIdx.x * 128, y0 = blockIdx.y * 64, b = blockIdx.z;
  v8f acc[4] = {};
  gemm_tile_128x64<true>(tmp + (size_t)b * Nseq * Lnum * PADW + (size_t)r0 * PADW,
                         PADW,
                         h1 + (size_t)(b * Nseq + y0) * PADW,
                         PADW, PADW, sA, sBT, acc);
  int roff = (lane & 16) ? 8 : 0;
#pragma unroll
  for (int si = 0; si < 2; ++si)
#pragma unroll
    for (int sj = 0; sj < 2; ++sj) {
      v8f cacc = acc[si * 2 + sj];
      int y = y0 + nw * 32 + sj * 16 + (lane & 15);
      int rowb = r0 + mw * 32 + si * 16 + roff;
#pragma unroll
      for (int vr = 0; vr < 8; ++vr) {
        int rr = rowb + vr;
        int x = rr >> 6, l = rr & 63;
        sflat[(((size_t)(b * Nseq + x)) * Nseq + y) * Lnum + l] = cacc[vr];
      }
    }
}

// -------------------- per-(m,t) logsumexp normalizer -----------------------
__global__ void k_row_lse(const float* __restrict__ sflat, float* __restrict__ norm) {
  int wave = threadIdx.x >> 5, lane = threadIdx.x & 31;
  size_t row = (size_t)blockIdx.x * 8 + wave;      // 1,048,576 rows
  const float* p = sflat + row * Lnum;
  float v0 = p[lane], v1 = p[lane + 32];
  float mx = fmaxf(v0, v1);
  for (int o = 16; o > 0; o >>= 1) mx = fmaxf(mx, __shfl_xor(mx, o, 32));
  float s = __expf(v0 - mx) + __expf(v1 - mx);
  for (int o = 16; o > 0; o >>= 1) s += __shfl_xor(s, o, 32);
  norm[row] = mx + __logf(s);
}

// ------------------------- CRF forward scan --------------------------------
__global__ void k_crf_forward(const float* __restrict__ sflat, const float* __restrict__ norm,
                              const int* __restrict__ lmask, const float* __restrict__ start_t,
                              const float* __restrict__ trans, const float* __restrict__ end_t,
                              float* __restrict__ den) {
  __shared__ float tr[Lnum * Lnum];
  __shared__ float aSh[4][72];
  __shared__ float red[4][64];
  int tid = threadIdx.x;
  for (int i = tid; i < Lnum * Lnum; i += 256) tr[i] = trans[i];
  int rl = tid >> 6, j = tid & 63;
  int m = blockIdx.x * 4 + rl, b = m >> 8;
  const float* srow = sflat + (size_t)m * Nseq * Lnum;
  float a = start_t[j] + (srow[j] - norm[(size_t)m * Nseq]);
  __syncthreads();
  aSh[rl][j] = a;
  for (int t = 1; t < Nseq; ++t) {
    __syncthreads();
    float mx = -3.0e38f;
#pragma unroll 8
    for (int i = 0; i < Lnum; ++i) mx = fmaxf(mx, aSh[rl][i] + tr[i * Lnum + j]);
    float s = 0.f;
#pragma unroll 8
    for (int i = 0; i < Lnum; ++i) s += __expf(aSh[rl][i] + tr[i * Lnum + j] - mx);
    float e = srow[(size_t)t * Lnum + j] - norm[(size_t)m * Nseq + t];
    float nxt = mx + __logf(s) + e;
    if (lmask[b * Nseq + t]) a = nxt;
    __syncthreads();
    aSh[rl][j] = a;
  }
  red[rl][j] = a + end_t[j];
  __syncthreads();
  if (j == 0) {
    float mx = red[rl][0];
    for (int i = 1; i < Lnum; ++i) mx = fmaxf(mx, red[rl][i]);
    float s = 0.f;
    for (int i = 0; i < Lnum; ++i) s += __expf(red[rl][i] - mx);
    den[m] = mx + __logf(s);
  }
}

// ------------------------- CRF numerator + weight --------------------------
__global__ void k_numerator(const float* __restrict__ sflat, const float* __restrict__ norm,
                            const int* __restrict__ lmask, const int* __restrict__ labels,
                            const float* __restrict__ start_t, const float* __restrict__ trans,
                            const float* __restrict__ end_t, const float* __restrict__ den,
                            float* __restrict__ contrib) {
  int m = blockIdx.x * 256 + threadIdx.x;          // 4096
  int b = m >> 8;
  const int* tg = labels + (size_t)m * Nseq;
  const float* srow = sflat + (size_t)m * Nseq * Lnum;
  int prev = tg[0];
  float num = start_t[prev];
  int cnt = 0;
  for (int t = 0; t < Nseq; ++t) {
    int tag = tg[t];
    int mk = lmask[b * Nseq + t];
    float mf = mk ? 1.f : 0.f;
    cnt += mk ? 1 : 0;
    num += (srow[(size_t)t * Lnum + tag] - norm[(size_t)m * Nseq + t]) * mf;
    if (t > 0) num += trans[prev * Lnum + tag] * mf;
    prev = tag;
  }
  num += end_t[tg[cnt - 1]];
  float w = lmask[m] ? 1.f : 0.f;                  // row_w = logits_mask.flat[m]
  contrib[m] = (num - den[m]) * w;
}

// deterministic fixed-order loss reduction
__global__ void k_reduce_loss(const float* __restrict__ contrib, float* __restrict__ out0) {
  __shared__ float sh[256];
  float s = 0.f;
  for (int i = threadIdx.x; i < MROWS; i += 256) s += contrib[i];
  sh[threadIdx.x] = s;
  __syncthreads();
  for (int st = 128; st > 0; st >>= 1) {
    if (threadIdx.x < st) sh[threadIdx.x] += sh[threadIdx.x + st];
    __syncthreads();
  }
  if (threadIdx.x == 0) out0[0] = -sh[0];
}

// ------------------------- Viterbi forward scan ----------------------------
__global__ void k_vit_forward(const float* __restrict__ sflat, const float* __restrict__ norm,
                              const int* __restrict__ lmask, const float* __restrict__ start_t,
                              const float* __restrict__ trans, const float* __restrict__ end_t,
                              u8* __restrict__ hist, int* __restrict__ last) {
  __shared__ float tr[Lnum * Lnum];
  __shared__ float aSh[4][72];
  __shared__ float red[4][64];
  int tid = threadIdx.x;
  for (int i = tid; i < Lnum * Lnum; i += 256) tr[i] = trans[i];
  int rl = tid >> 6, j = tid & 63;
  int m = blockIdx.x * 4 + rl, b = m >> 8;
  const float* srow = sflat + (size_t)m * Nseq * Lnum;
  float a = start_t[j] + (srow[j] - norm[(size_t)m * Nseq]);
  __syncthreads();
  aSh[rl][j] = a;
  for (int t = 1; t < Nseq; ++t) {
    __syncthreads();
    float best = aSh[rl][0] + tr[j];
    int bi = 0;
#pragma unroll 8
    for (int i = 1; i < Lnum; ++i) {
      float v = aSh[rl][i] + tr[i * Lnum + j];
      if (v > best) { best = v; bi = i; }          // first-max (jnp.argmax)
    }
    float e = srow[(size_t)t * Lnum + j] - norm[(size_t)m * Nseq + t];
    int mk = lmask[b * Nseq + t];
    hist[((size_t)(t - 1) * MROWS + m) * Lnum + j] = (u8)(mk ? bi : j);
    if (mk) a = best + e;
    __syncthreads();
    aSh[rl][j] = a;
  }
  red[rl][j] = a + end_t[j];
  __syncthreads();
  if (j == 0) {
    float best = red[rl][0];
    int bi = 0;
    for (int i = 1; i < Lnum; ++i)
      if (red[rl][i] > best) { best = red[rl][i]; bi = i; }
    last[m] = bi;
  }
}

__global__ void k_backtrack(const u8* __restrict__ hist, const int* __restrict__ last,
                            float* __restrict__ decoded) {
  int m = blockIdx.x * 256 + threadIdx.x;          // 4096
  int cur = last[m];
  decoded[(size_t)m * Nseq + (Nseq - 1)] = (float)cur;
  for (int t = Nseq - 2; t >= 0; --t) {
    cur = hist[((size_t)t * MROWS + m) * Lnum + cur];
    decoded[(size_t)m * Nseq + t] = (float)cur;
  }
}

// ---------------------------------------------------------------------------
extern "C" void kernel_launch(void* const* d_in, const int* in_sizes, int n_in,
                              void* d_out, int out_size, void* d_ws, size_t ws_size,
                              hipStream_t stream) {
  (void)in_sizes; (void)n_in; (void)out_size; (void)ws_size;
  const float* in_f    = (const float*)d_in[0];
  const int*   lmask   = (const int*)d_in[1];
  const int*   labels  = (const int*)d_in[2];
  const float* Wh      = (const float*)d_in[3];
  const float* bh      = (const float*)d_in[4];
  const float* Wd      = (const float*)d_in[5];
  const float* bd      = (const float*)d_in[6];
  const float* bw      = (const float*)d_in[7];
  const float* start_t = (const float*)d_in[8];
  const float* trans   = (const float*)d_in[9];
  const float* end_t   = (const float*)d_in[10];

  char* ws = (char*)d_ws;
  size_t off = 0;
  auto bump = [&](size_t bytes) { void* p = ws + off; off += (bytes + 255) & ~(size_t)255; return p; };

  _Float16* inF16 = (_Float16*)bump((size_t)MROWS * Hdim * 2);
  _Float16* bmatT = (_Float16*)bump((size_t)2 * PADW * Hdim * 2);
  float*    bias  = (float*)   bump((size_t)2 * PADW * 4);
  _Float16* h1    = (_Float16*)bump((size_t)MROWS * PADW * 2);
  _Float16* d1    = (_Float16*)bump((size_t)MROWS * PADW * 2);
  _Float16* wpT   = (_Float16*)bump((size_t)Lnum * PADW * PADW * 2);
  _Float16* tmp   = (_Float16*)bump((size_t)MROWS * Lnum * PADW * 2);   // 168 MB
  float*    norm  = (float*)   bump((size_t)MROWS * Nseq * 4);
  float*    den   = (float*)   bump((size_t)MROWS * 4);
  float*    contr = (float*)   bump((size_t)MROWS * 4);
  int*      last  = (int*)     bump((size_t)MROWS * 4);
  u8*       hist  = (u8*)tmp;   // reuse tmp region after bilin2 (67 MB < 168 MB)

  float* out     = (float*)d_out;
  float* sflat   = out + 1;                                   // 4096*256*64
  float* decoded = out + 1 + (size_t)MROWS * Nseq * Lnum;     // 4096*256

  // prep
  k_cast_input<<<(MROWS * Hdim) / 256, 256, 0, stream>>>(in_f, inF16, MROWS * Hdim);
  k_prep_bmatT<<<(2 * PADW * Hdim) / 256, 256, 0, stream>>>(Wh, bh, Wd, bd, bmatT, bias);
  k_prep_wpT<<<(Lnum * PADW * PADW) / 256, 256, 0, stream>>>(bw, wpT);

  // WMMA GEMMs (128x64 workgroup tile, 4 WMMAs/wave/K-step)
  k_mlp_gemm<<<dim3(MROWS / 128, PADW / 64, 2), 256, 0, stream>>>(inF16, bmatT, bias, h1, d1);
  k_bilin1 <<<dim3(MROWS / 128, PADW / 64, Lnum), 256, 0, stream>>>(d1, wpT, tmp);
  k_bilin2 <<<dim3((Nseq * Lnum) / 128, Nseq / 64, Bsz), 256, 0, stream>>>(tmp, h1, sflat);

  // CRF loss
  k_row_lse<<<(MROWS * Nseq) / 8, 256, 0, stream>>>(sflat, norm);
  k_crf_forward<<<MROWS / 4, 256, 0, stream>>>(sflat, norm, lmask, start_t, trans, end_t, den);
  k_numerator<<<MROWS / 256, 256, 0, stream>>>(sflat, norm, lmask, labels, start_t, trans, end_t, den, contr);
  k_reduce_loss<<<1, 256, 0, stream>>>(contr, out);

  // Viterbi decode (reuses tmp region for history)
  k_vit_forward<<<MROWS / 4, 256, 0, stream>>>(sflat, norm, lmask, start_t, trans, end_t, hist, last);
  k_backtrack<<<MROWS / 256, 256, 0, stream>>>(hist, last, decoded);
}
